// BigARDecoder_20916490732283
// MI455X (gfx1250) — matrix operand order repaired and checked by reading, not compiled, not run on hardware
//
#include <hip/hip_runtime.h>
#include <hip/hip_bf16.h>
#include <cstdint>
#include <cstddef>

// ---------------------------------------------------------------------------
// BigARDecoder for MI455X (gfx1250).
//  - Persistent-kernel design: 64 blocks x 256 threads (512 wave32 waves),
//    grid-wide barrier between dependent GEMM layers (sequential recurrence,
//    ~2800 dependent GEMM rounds -> per-layer kernel launches would dominate).
//  - All GEMMs use v_wmma_f32_16x16x32_bf16 (M=64 => 4 WMMA M-tiles).
//  - Weights converted once fp32->bf16 into workspace; ~100MB bf16 weights
//    stay resident in the 192MB L2 across all 255 steps.
//  - Cell state c kept in fp32; activations as bf16 for WMMA inputs.
//  - Transcendentals via v_exp_f32 + v_rcp_f32 (no IEEE divide chains on the
//    latency-critical elementwise passes).
//  - Independent GEMM pairs (h/c adapters, lastH/lastC) run on disjoint wave
//    partitions so both proceed concurrently within one barrier interval.
// ---------------------------------------------------------------------------

typedef __attribute__((ext_vector_type(16))) __bf16 v16bf;
typedef __attribute__((ext_vector_type(8)))  __bf16 v8bf;
typedef __attribute__((ext_vector_type(8)))  float  v8f;

#define NBLK 64
#define TPB  256
#define NWAVES (NBLK * (TPB / 32))   // 512 waves

// problem dims
static constexpr int BDIM = 64;      // batch
static constexpr int TDIM = 256;     // time
static constexpr int KIN  = 256;     // z features
static constexpr int HDIM = 1024;
static constexpr int GODIM = 320;    // G + O
static constexpr int DDEP = 4;
static constexpr int TSTEPS = TDIM - 1; // 255

// ---------------------------------------------------------------------------
// fast transcendentals (v_exp_f32 / v_rcp_f32)
// ---------------------------------------------------------------------------
__device__ inline float sigm(float x) {
  return __builtin_amdgcn_rcpf(1.f + __expf(-x));
}
__device__ inline float fast_tanh(float x) {
  // tanh(x) = 1 - 2/(exp(2x)+1); exp overflow -> inf -> rcp -> 0 -> 1 (correct)
  return 1.f - 2.f * __builtin_amdgcn_rcpf(__expf(2.f * x) + 1.f);
}

// ---------------------------------------------------------------------------
// grid-wide barrier (monotone counter; all 64 blocks resident on MI455X)
// ---------------------------------------------------------------------------
__device__ inline void grid_barrier(unsigned* bar, unsigned target) {
  __syncthreads();
  if (threadIdx.x == 0) {
    __threadfence();
    __hip_atomic_fetch_add(bar, 1u, __ATOMIC_ACQ_REL, __HIP_MEMORY_SCOPE_AGENT);
    while (__hip_atomic_load(bar, __ATOMIC_ACQUIRE, __HIP_MEMORY_SCOPE_AGENT) < target) {
      __builtin_amdgcn_s_sleep(1);
    }
    __threadfence();
  }
  __syncthreads();
}

// ---------------------------------------------------------------------------
// K-loop for one 16x16 output tile: acc += A[mb:mb+16, :K] @ W[nb:nb+16, :K]^T
// A is [64,K] row-major bf16 (lda elems). W is [N,K] row-major bf16 (ld=K).
// bf16 fragment k order used by this code (same for A and B; any matching order works):
//   A (16x32): lane&15 = M; halves 0..15 = K(h*16 .. h*16+15)  (contiguous)
//   B (32x16): lane&15 = N; halves 0..15 = K(h*16 .. h*16+15)  (contiguous!)
// ---------------------------------------------------------------------------
__device__ inline v8f ktile_loop(const __bf16* __restrict__ A, int lda,
                                 const __bf16* __restrict__ W, int K,
                                 int mb, int nb, int lane, v8f acc) {
  const int h = lane >> 4;
  const __bf16* ar = A + (size_t)(mb + (lane & 15)) * lda + h * 16;
  const __bf16* wr = W + (size_t)(nb + (lane & 15)) * K   + h * 16;
#pragma unroll 2
  for (int k = 0; k < K; k += 32) {
    v8bf  alo = *(const v8bf*)(ar + k);
    v8bf  ahi = *(const v8bf*)(ar + k + 8);
    v16bf bfr = *(const v16bf*)(wr + k);
    v16bf afr;
#pragma unroll
    for (int i = 0; i < 8; ++i) { afr[i] = alo[i]; afr[i + 8] = ahi[i]; }
    acc = __builtin_amdgcn_wmma_f32_16x16x32_bf16(
        /*neg_a=*/false, afr, /*neg_b=*/false, bfr,
        /*c_mod=*/(short)0, acc, /*reuse_a=*/false, /*reuse_b=*/false);
  }
  return acc;
}

// Generic dual-input GEMM pass:  out = act(A1@W1^T + A2@W2^T + bias)
// Output C/D layout: lane&15 = n; VGPR r -> m = (lane>>4)*8 + r.
// Tiles are distributed over waves [wbase, wbase+wstride) so two independent
// GEMMs can share one barrier interval on disjoint wave partitions.
__device__ inline void gemm(const __bf16* A1, int lda1, const __bf16* W1, int K1,
                            const __bf16* A2, int lda2, const __bf16* W2, int K2,
                            const float* bias, int N, int act,
                            float* oF, int ldF, __bf16* oB, int ldB,
                            int wave, int lane, int wbase = 0, int wstride = NWAVES) {
  const int w = wave - wbase;
  if (w < 0 || w >= wstride) return;
  const int ntiles = 4 * (N >> 4);
  for (int t = w; t < ntiles; t += wstride) {
    const int mb = (t & 3) << 4;
    const int nb = (t >> 2) << 4;
    v8f acc = {0.f, 0.f, 0.f, 0.f, 0.f, 0.f, 0.f, 0.f};
    acc = ktile_loop(A1, lda1, W1, K1, mb, nb, lane, acc);
    if (A2) acc = ktile_loop(A2, lda2, W2, K2, mb, nb, lane, acc);
    const int n  = nb + (lane & 15);
    const int m0 = mb + (lane >> 4) * 8;
    const float bv = bias ? bias[n] : 0.f;
#pragma unroll
    for (int r = 0; r < 8; ++r) {
      float v = acc[r] + bv;
      if (act) v = v * sigm(v);                      // SiLU via v_rcp_f32
      if (oF) oF[(size_t)(m0 + r) * ldF + n] = v;
      if (oB) oB[(size_t)(m0 + r) * ldB + n] = (__bf16)v;
    }
  }
}

// ---------------------------------------------------------------------------
// LSTM elementwise pass over gates [64, 4*Ng] (i,f,g,o chunk order)
// ---------------------------------------------------------------------------
__device__ inline void lstm_elem(const float* gates, int Ng, const float* b4,
                                 const float* cprev, int ldcp,
                                 __bf16* hB, int ldhB,
                                 float* cF, int ldcF, __bf16* cB, int ldcB,
                                 float* hF, size_t ldhF, int tid) {
  const int total = BDIM * Ng;
  const int ldg = 4 * Ng;
  for (int idx = tid; idx < total; idx += NBLK * TPB) {
    const int b = idx / Ng, n = idx - b * Ng;
    const float* g = gates + (size_t)b * ldg;
    const float gi = g[n]          + b4[n];
    const float gf = g[Ng + n]     + b4[Ng + n];
    const float gg = g[2 * Ng + n] + b4[2 * Ng + n];
    const float go = g[3 * Ng + n] + b4[3 * Ng + n];
    const float c  = sigm(gf) * cprev[(size_t)b * ldcp + n] + sigm(gi) * fast_tanh(gg);
    const float h  = sigm(go) * fast_tanh(c);
    if (cF) cF[(size_t)b * ldcF + n] = c;
    if (cB) cB[(size_t)b * ldcB + n] = (__bf16)c;
    if (hB) hB[(size_t)b * ldhB + n] = (__bf16)h;
    if (hF) hF[(size_t)b * ldhF + n] = h;
  }
}

// ---------------------------------------------------------------------------
// Parameter block
// ---------------------------------------------------------------------------
struct DecParams {
  // bf16 weights (workspace copies)
  const __bf16 *ehW1, *ehW2, *ehW3;
  const __bf16 *ecW1, *ecW2, *ecW3;
  const __bf16 *exW1, *exW2, *exW3;
  const __bf16 *inWih, *inWhh;
  const __bf16 *rWih, *rWhh, *hAW, *cAW;
  const __bf16 *lHW, *lCW, *oWih, *oWhh;
  // fp32 biases (original inputs / precomputed sums)
  const float *eh_b1, *eh_b2, *eh_b3;
  const float *ec_b1, *ec_b2, *ec_b3;
  const float *ex_b1, *ex_b2, *ex_b3;
  const float *hA_b, *cA_b, *lH_b, *lC_b;
  const float *bsum_in, *bsum_r, *bsum_out;
  // activations / state
  const __bf16 *zb, *mt;
  __bf16 *t1, *t2, *h2b, *h1b, *c2b, *c1b, *hhb;
  float *c2f, *c1f, *ccf, *gates;
  float *out;          // d_out [64,256,320] fp32
  unsigned *bar;
};

// ---------------------------------------------------------------------------
// Persistent decoder kernel
// ---------------------------------------------------------------------------
__global__ __launch_bounds__(TPB) void big_ar_decoder(DecParams p) {
  const int lane = threadIdx.x & 31;
  const int wave = blockIdx.x * (TPB / 32) + (threadIdx.x >> 5);
  const int tid  = blockIdx.x * TPB + threadIdx.x;
  unsigned tgt = 0;
#define GS() do { tgt += NBLK; grid_barrier(p.bar, tgt); } while (0)

  // ---- encoder MLPs (Linear->SiLU->Linear->SiLU->Linear) ----
  // h0 = mlp_eh(z) -> h2b (bf16)
  gemm(p.zb, KIN, p.ehW1, KIN, nullptr, 0, nullptr, 0, p.eh_b1, HDIM, 1,
       nullptr, 0, p.t1, HDIM, wave, lane); GS();
  gemm(p.t1, HDIM, p.ehW2, HDIM, nullptr, 0, nullptr, 0, p.eh_b2, HDIM, 1,
       nullptr, 0, p.t2, HDIM, wave, lane); GS();
  gemm(p.t2, HDIM, p.ehW3, HDIM, nullptr, 0, nullptr, 0, p.eh_b3, HDIM, 0,
       nullptr, 0, p.h2b, HDIM, wave, lane); GS();
  // c0 = mlp_ec(z) -> c2f (fp32)
  gemm(p.zb, KIN, p.ecW1, KIN, nullptr, 0, nullptr, 0, p.ec_b1, HDIM, 1,
       nullptr, 0, p.t1, HDIM, wave, lane); GS();
  gemm(p.t1, HDIM, p.ecW2, HDIM, nullptr, 0, nullptr, 0, p.ec_b2, HDIM, 1,
       nullptr, 0, p.t2, HDIM, wave, lane); GS();
  gemm(p.t2, HDIM, p.ecW3, HDIM, nullptr, 0, nullptr, 0, p.ec_b3, HDIM, 0,
       p.c2f, HDIM, nullptr, 0, wave, lane); GS();
  // x_last = mlp_ex(z) -> d_out[:,0,256:320] (cols 0:256 zeroed in prep)
  gemm(p.zb, KIN, p.exW1, KIN, nullptr, 0, nullptr, 0, p.ex_b1, HDIM, 1,
       nullptr, 0, p.t1, HDIM, wave, lane); GS();
  gemm(p.t1, HDIM, p.exW2, HDIM, nullptr, 0, nullptr, 0, p.ex_b2, HDIM, 1,
       nullptr, 0, p.t2, HDIM, wave, lane); GS();
  gemm(p.t2, HDIM, p.exW3, HDIM, nullptr, 0, nullptr, 0, p.ex_b3, 64, 0,
       p.out + 256, TDIM * GODIM, nullptr, 0, wave, lane); GS();

  // ---- autoregressive scan over 255 steps ----
  for (int t = 0; t < TSTEPS; ++t) {
    const __bf16* x = p.mt + (size_t)t * BDIM * GODIM;

    // in-cell gates: x @ in_Wih^T + h @ in_Whh^T
    gemm(x, GODIM, p.inWih, GODIM, p.h2b, HDIM, p.inWhh, HDIM,
         nullptr, 4 * HDIM, 0, p.gates, 4 * HDIM, nullptr, 0, wave, lane); GS();
    lstm_elem(p.gates, HDIM, p.bsum_in, p.c2f, HDIM,
              p.h2b, HDIM, p.c2f, HDIM, p.c2b, HDIM, nullptr, 0, tid); GS();

    // inner recurrent chain
    for (int i = 0; i < DDEP; ++i) {
      const __bf16* Wih = p.rWih + (size_t)i * 4 * HDIM * HDIM;
      const __bf16* Whh = p.rWhh + (size_t)i * 4 * HDIM * HDIM;
      const __bf16* hA  = p.hAW  + (size_t)i * HDIM * HDIM;
      const __bf16* cA  = p.cAW  + (size_t)i * HDIM * HDIM;
      // adapters on disjoint wave halves: h1 = h2@hA^T + b ; c1 = c2@cA^T + b
      gemm(p.h2b, HDIM, hA, HDIM, nullptr, 0, nullptr, 0,
           p.hA_b + i * HDIM, HDIM, 0, nullptr, 0, p.h1b, HDIM,
           wave, lane, 0, NWAVES / 2);
      gemm(p.c2b, HDIM, cA, HDIM, nullptr, 0, nullptr, 0,
           p.cA_b + i * HDIM, HDIM, 0, p.c1f, HDIM, p.c1b, HDIM,
           wave, lane, NWAVES / 2, NWAVES / 2); GS();
      // gates: h2@Wih^T + h1@Whh^T
      gemm(p.h2b, HDIM, Wih, HDIM, p.h1b, HDIM, Whh, HDIM,
           nullptr, 4 * HDIM, 0, p.gates, 4 * HDIM, nullptr, 0, wave, lane); GS();
      lstm_elem(p.gates, HDIM, p.bsum_r + i * 4 * HDIM, p.c1f, HDIM,
                p.h2b, HDIM, p.c2f, HDIM, p.c2b, HDIM, nullptr, 0, tid); GS();
    }

    // hh = h2@lastH^T + b ; cc = c1@lastC^T + b  (disjoint wave halves)
    gemm(p.h2b, HDIM, p.lHW, HDIM, nullptr, 0, nullptr, 0,
         p.lH_b, GODIM, 0, nullptr, 0, p.hhb, GODIM,
         wave, lane, 0, NWAVES / 2);
    gemm(p.c1b, HDIM, p.lCW, HDIM, nullptr, 0, nullptr, 0,
         p.lC_b, GODIM, 0, p.ccf, GODIM, nullptr, 0,
         wave, lane, NWAVES / 2, NWAVES / 2); GS();
    // out-cell gates: h2@out_Wih^T + hh@out_Whh^T
    gemm(p.h2b, HDIM, p.oWih, HDIM, p.hhb, GODIM, p.oWhh, GODIM,
         nullptr, 4 * GODIM, 0, p.gates, 4 * GODIM, nullptr, 0, wave, lane); GS();
    // ho -> m_hat[:, t+1, :]
    lstm_elem(p.gates, GODIM, p.bsum_out, p.ccf, GODIM,
              nullptr, 0, nullptr, 0, nullptr, 0,
              p.out + (size_t)(t + 1) * GODIM, (size_t)TDIM * GODIM, tid); GS();
  }
#undef GS
}

// ---------------------------------------------------------------------------
// prep kernels
// ---------------------------------------------------------------------------
__global__ void cvt_kernel(const float* __restrict__ s, __bf16* __restrict__ d, int n) {
  int i = blockIdx.x * blockDim.x + threadIdx.x;
  if (i < n) d[i] = (__bf16)s[i];
}
__global__ void addb_kernel(const float* __restrict__ a, const float* __restrict__ b,
                            float* __restrict__ d, int n) {
  int i = blockIdx.x * blockDim.x + threadIdx.x;
  if (i < n) d[i] = a[i] + b[i];
}
// mt[j,b,c] = bf16(m_true[b, T-1-j, c]),  j in [0,255)
__global__ void build_mt_kernel(const float* __restrict__ m_true, __bf16* __restrict__ mt) {
  int i = blockIdx.x * blockDim.x + threadIdx.x;
  const int n = TSTEPS * BDIM * GODIM;
  if (i >= n) return;
  int j = i / (BDIM * GODIM);
  int r = i - j * (BDIM * GODIM);
  int b = r / GODIM, c = r - b * GODIM;
  mt[i] = (__bf16)m_true[(size_t)b * TDIM * GODIM + (size_t)(TDIM - 1 - j) * GODIM + c];
}
// zero graph part of m_hat[:,0,0:256]
__global__ void zero_row0_kernel(float* out) {
  int i = blockIdx.x * blockDim.x + threadIdx.x;
  if (i >= BDIM * 256) return;
  int b = i / 256, n = i - b * 256;
  out[(size_t)b * TDIM * GODIM + n] = 0.f;
}

// ---------------------------------------------------------------------------
// host launcher
// ---------------------------------------------------------------------------
static inline void cvt_launch(const float* s, __bf16* d, int n, hipStream_t st) {
  hipLaunchKernelGGL(cvt_kernel, dim3((n + TPB - 1) / TPB), dim3(TPB), 0, st, s, d, n);
}

extern "C" void kernel_launch(void* const* d_in, const int* in_sizes, int n_in,
                              void* d_out, int out_size, void* d_ws, size_t ws_size,
                              hipStream_t stream) {
  (void)in_sizes; (void)n_in; (void)out_size; (void)ws_size;
  const float* z      = (const float*)d_in[0];
  const float* m_true = (const float*)d_in[1];
  const float* ehW1 = (const float*)d_in[2];  const float* eh_b1 = (const float*)d_in[3];
  const float* ehW2 = (const float*)d_in[4];  const float* eh_b2 = (const float*)d_in[5];
  const float* ehW3 = (const float*)d_in[6];  const float* eh_b3 = (const float*)d_in[7];
  const float* ecW1 = (const float*)d_in[8];  const float* ec_b1 = (const float*)d_in[9];
  const float* ecW2 = (const float*)d_in[10]; const float* ec_b2 = (const float*)d_in[11];
  const float* ecW3 = (const float*)d_in[12]; const float* ec_b3 = (const float*)d_in[13];
  const float* exW1 = (const float*)d_in[14]; const float* ex_b1 = (const float*)d_in[15];
  const float* exW2 = (const float*)d_in[16]; const float* ex_b2 = (const float*)d_in[17];
  const float* exW3 = (const float*)d_in[18]; const float* ex_b3 = (const float*)d_in[19];
  const float* inWih = (const float*)d_in[20]; const float* inWhh = (const float*)d_in[21];
  const float* in_bih = (const float*)d_in[22]; const float* in_bhh = (const float*)d_in[23];
  const float* rWih = (const float*)d_in[24]; const float* rWhh = (const float*)d_in[25];
  const float* r_bih = (const float*)d_in[26]; const float* r_bhh = (const float*)d_in[27];
  const float* hAW = (const float*)d_in[28]; const float* hA_b = (const float*)d_in[29];
  const float* cAW = (const float*)d_in[30]; const float* cA_b = (const float*)d_in[31];
  const float* lHW = (const float*)d_in[32]; const float* lH_b = (const float*)d_in[33];
  const float* lCW = (const float*)d_in[34]; const float* lC_b = (const float*)d_in[35];
  const float* oWih = (const float*)d_in[36]; const float* oWhh = (const float*)d_in[37];
  const float* o_bih = (const float*)d_in[38]; const float* o_bhh = (const float*)d_in[39];

  char* base = (char*)d_ws;
  size_t off = 0;
  auto ab = [&](size_t n) -> __bf16* {   // bf16 alloc
    void* r = base + off; off = (off + n * 2 + 255) & ~(size_t)255; return (__bf16*)r;
  };
  auto af = [&](size_t n) -> float* {    // f32 alloc
    void* r = base + off; off = (off + n * 4 + 255) & ~(size_t)255; return (float*)r;
  };

  DecParams p;
  // bf16 weight copies
  __bf16* w_ehW1 = ab(HDIM * KIN);    __bf16* w_ehW2 = ab(HDIM * HDIM);  __bf16* w_ehW3 = ab(HDIM * HDIM);
  __bf16* w_ecW1 = ab(HDIM * KIN);    __bf16* w_ecW2 = ab(HDIM * HDIM);  __bf16* w_ecW3 = ab(HDIM * HDIM);
  __bf16* w_exW1 = ab(HDIM * KIN);    __bf16* w_exW2 = ab(HDIM * HDIM);  __bf16* w_exW3 = ab(64 * HDIM);
  __bf16* w_inWih = ab(4 * HDIM * GODIM);          __bf16* w_inWhh = ab(4 * HDIM * HDIM);
  __bf16* w_rWih = ab((size_t)DDEP * 4 * HDIM * HDIM);
  __bf16* w_rWhh = ab((size_t)DDEP * 4 * HDIM * HDIM);
  __bf16* w_hAW  = ab((size_t)DDEP * HDIM * HDIM);
  __bf16* w_cAW  = ab((size_t)DDEP * HDIM * HDIM);
  __bf16* w_lHW  = ab(GODIM * HDIM); __bf16* w_lCW = ab(GODIM * HDIM);
  __bf16* w_oWih = ab(4 * GODIM * HDIM); __bf16* w_oWhh = ab(4 * GODIM * GODIM);
  // activations / state
  __bf16* zb = ab(BDIM * KIN);
  __bf16* mt = ab((size_t)TSTEPS * BDIM * GODIM);
  __bf16* t1 = ab(BDIM * HDIM);  __bf16* t2 = ab(BDIM * HDIM);
  __bf16* h2b = ab(BDIM * HDIM); __bf16* h1b = ab(BDIM * HDIM);
  __bf16* c2b = ab(BDIM * HDIM); __bf16* c1b = ab(BDIM * HDIM);
  __bf16* hhb = ab(BDIM * GODIM);
  float* c2f = af(BDIM * HDIM);  float* c1f = af(BDIM * HDIM);
  float* ccf = af(BDIM * GODIM); float* gates = af(BDIM * 4 * HDIM);
  float* bsum_in = af(4 * HDIM);
  float* bsum_r  = af(DDEP * 4 * HDIM);
  float* bsum_out = af(4 * GODIM);
  unsigned* bar = (unsigned*)(base + off); off += 256;

  // --- prep: barrier reset, conversions, bias sums, mt, row-0 zeros ---
  hipMemsetAsync(bar, 0, 256, stream);
  cvt_launch(ehW1, w_ehW1, HDIM * KIN, stream);
  cvt_launch(ehW2, w_ehW2, HDIM * HDIM, stream);
  cvt_launch(ehW3, w_ehW3, HDIM * HDIM, stream);
  cvt_launch(ecW1, w_ecW1, HDIM * KIN, stream);
  cvt_launch(ecW2, w_ecW2, HDIM * HDIM, stream);
  cvt_launch(ecW3, w_ecW3, HDIM * HDIM, stream);
  cvt_launch(exW1, w_exW1, HDIM * KIN, stream);
  cvt_launch(exW2, w_exW2, HDIM * HDIM, stream);
  cvt_launch(exW3, w_exW3, 64 * HDIM, stream);
  cvt_launch(inWih, w_inWih, 4 * HDIM * GODIM, stream);
  cvt_launch(inWhh, w_inWhh, 4 * HDIM * HDIM, stream);
  cvt_launch(rWih, w_rWih, DDEP * 4 * HDIM * HDIM, stream);
  cvt_launch(rWhh, w_rWhh, DDEP * 4 * HDIM * HDIM, stream);
  cvt_launch(hAW, w_hAW, DDEP * HDIM * HDIM, stream);
  cvt_launch(cAW, w_cAW, DDEP * HDIM * HDIM, stream);
  cvt_launch(lHW, w_lHW, GODIM * HDIM, stream);
  cvt_launch(lCW, w_lCW, GODIM * HDIM, stream);
  cvt_launch(oWih, w_oWih, 4 * GODIM * HDIM, stream);
  cvt_launch(oWhh, w_oWhh, 4 * GODIM * GODIM, stream);
  cvt_launch(z, zb, BDIM * KIN, stream);
  hipLaunchKernelGGL(addb_kernel, dim3((4 * HDIM + TPB - 1) / TPB), dim3(TPB), 0, stream,
                     in_bih, in_bhh, bsum_in, 4 * HDIM);
  hipLaunchKernelGGL(addb_kernel, dim3((DDEP * 4 * HDIM + TPB - 1) / TPB), dim3(TPB), 0, stream,
                     r_bih, r_bhh, bsum_r, DDEP * 4 * HDIM);
  hipLaunchKernelGGL(addb_kernel, dim3((4 * GODIM + TPB - 1) / TPB), dim3(TPB), 0, stream,
                     o_bih, o_bhh, bsum_out, 4 * GODIM);
  {
    const int n = TSTEPS * BDIM * GODIM;
    hipLaunchKernelGGL(build_mt_kernel, dim3((n + TPB - 1) / TPB), dim3(TPB), 0, stream, m_true, mt);
  }
  hipLaunchKernelGGL(zero_row0_kernel, dim3((BDIM * 256 + TPB - 1) / TPB), dim3(TPB), 0, stream,
                     (float*)d_out);

  // --- persistent decoder ---
  p.ehW1 = w_ehW1; p.ehW2 = w_ehW2; p.ehW3 = w_ehW3;
  p.ecW1 = w_ecW1; p.ecW2 = w_ecW2; p.ecW3 = w_ecW3;
  p.exW1 = w_exW1; p.exW2 = w_exW2; p.exW3 = w_exW3;
  p.inWih = w_inWih; p.inWhh = w_inWhh;
  p.rWih = w_rWih; p.rWhh = w_rWhh; p.hAW = w_hAW; p.cAW = w_cAW;
  p.lHW = w_lHW; p.lCW = w_lCW; p.oWih = w_oWih; p.oWhh = w_oWhh;
  p.eh_b1 = eh_b1; p.eh_b2 = eh_b2; p.eh_b3 = eh_b3;
  p.ec_b1 = ec_b1; p.ec_b2 = ec_b2; p.ec_b3 = ec_b3;
  p.ex_b1 = ex_b1; p.ex_b2 = ex_b2; p.ex_b3 = ex_b3;
  p.hA_b = hA_b; p.cA_b = cA_b; p.lH_b = lH_b; p.lC_b = lC_b;
  p.bsum_in = bsum_in; p.bsum_r = bsum_r; p.bsum_out = bsum_out;
  p.zb = zb; p.mt = mt;
  p.t1 = t1; p.t2 = t2; p.h2b = h2b; p.h1b = h1b; p.c2b = c2b; p.c1b = c1b; p.hhb = hhb;
  p.c2f = c2f; p.c1f = c1f; p.ccf = ccf; p.gates = gates;
  p.out = (float*)d_out; p.bar = bar;

  hipLaunchKernelGGL(big_ar_decoder, dim3(NBLK), dim3(TPB), 0, stream, p);
}
